// RCNN_34402688041114
// MI455X (gfx1250) — matrix operand order repaired and checked by reading, not compiled, not run on hardware
//
#include <hip/hip_runtime.h>

typedef __attribute__((ext_vector_type(16))) _Float16 v16h;
typedef __attribute__((ext_vector_type(8)))  float    v8f;

#define NTT   512          // T
#define K1S   3            // conv1 k-steps (K=96 padded from 72)
#define K2    1536         // conv2 K = 256*6
#define K2S   48           // conv2 k-steps

#define WMMA_F16(A, B, C) __builtin_amdgcn_wmma_f32_16x16x32_f16( \
        false, (A), false, (B), (short)0, (C), false, false)

// 16-bit A-matrix 16x32 register layout, contiguous k order (matches B packing):
// lane half (lane/16) and in-lane half index ii (0..15) -> kl = 16*half + ii in [0,32)
__device__ __forceinline__ int a_kl(int ii, int half) {
    return ii + 16 * half;
}

// Pack conv1 weights (256,2,3,4,3) f32 -> A-fragment f16, K padded 72->96.
// kpad = c*48 + kd*16 + (kh*3+kw), positions q in [12,16) are zero.
__global__ void pack_w1(const float* __restrict__ w1, _Float16* __restrict__ a1) {
    int idx  = blockIdx.x * blockDim.x + threadIdx.x;   // 24576 total
    int ii   = idx & 15;
    int lane = (idx >> 4) & 31;
    int ks   = (idx >> 9) % K1S;
    int mt   = idx / (K1S * 512);
    int m    = mt * 16 + (lane & 15);
    int kl   = a_kl(ii, lane >> 4);
    int k    = ks * 32 + kl;            // [0,96)
    int c    = k / 48;
    int rem  = k % 48;
    int kd   = rem >> 4;
    int q    = rem & 15;
    float v  = 0.f;
    if (q < 12) {
        int kh = q / 3, kw = q % 3;
        v = w1[(((m * 2 + c) * 3 + kd) * 4 + kh) * 3 + kw];
    }
    a1[idx] = (_Float16)v;
}

// Pack conv2 weights (256,256,1,6,1) f32 -> A-fragment f16; K = cin*6 + j.
__global__ void pack_w2(const float* __restrict__ w2, _Float16* __restrict__ a2) {
    int idx  = blockIdx.x * blockDim.x + threadIdx.x;   // 393216 total
    int ii   = idx & 15;
    int lane = (idx >> 4) & 31;
    int ks   = (idx >> 9) % K2S;
    int mt   = idx / (K2S * 512);
    int m    = mt * 16 + (lane & 15);
    int kl   = a_kl(ii, lane >> 4);
    int k    = ks * 32 + kl;            // [0,1536)
    a2[idx] = (_Float16)w2[m * K2 + k];
}

// conv1 as WMMA GEMM (M=256, K=96, N-tile=16 columns: fixed j, 16 consecutive t),
// fused bias+ReLU, writes result f16 directly in GEMM2's B-fragment layout.
__global__ void __launch_bounds__(256) conv1_wmma(
        const float* __restrict__ x, const _Float16* __restrict__ a1,
        const float* __restrict__ bias1, _Float16* __restrict__ bpk) {
    __shared__ alignas(32) _Float16 Bs[K1S * 512];   // [ks][lane][ii], B-frag layout
    int tid   = threadIdx.x;
    int bid   = blockIdx.x;           // 6144 = 6 j * 1024 tiles
    int j     = bid >> 10;            // 0..5
    int ntile = bid & 1023;           // tile over n = b*512 + t
    int b     = ntile >> 5;
    int t0    = (ntile & 31) << 4;

    // im2col patch -> LDS in B-fragment layout (lanes 0-15: K0-15, 16-31: K16-31)
    for (int idx = tid; idx < K1S * 512; idx += 256) {
        int ii   = idx & 15;
        int lane = (idx >> 4) & 31;
        int ks   = idx >> 9;
        int kl   = ii + ((lane >> 4) << 4);
        int k    = ks * 32 + kl;
        int c    = k / 48;
        int rem  = k % 48;
        int kd   = rem >> 4;
        int q    = rem & 15;
        int t    = t0 + (lane & 15);
        float v  = 0.f;
        if (q < 12) {
            int hh = 4 * j + q / 3 - 3;      // H pos with pad 3, stride 4
            int td = t + kd - 1;             // D pos with pad 1
            if (hh >= 0 && hh < 21 && td >= 0 && td < NTT)
                v = x[b * 64512 + c * 32256 + td * 63 + hh * 3 + (q % 3)];
        }
        Bs[idx] = (_Float16)v;
    }
    __syncthreads();

    int wave = tid >> 5, lane = tid & 31;
    #pragma unroll
    for (int mt2 = 0; mt2 < 2; ++mt2) {
        int mt = wave * 2 + mt2;
        v8f acc = {};
        #pragma unroll
        for (int ks = 0; ks < K1S; ++ks) {
            v16h af = *(const v16h*)(a1 + ((mt * K1S + ks) * 32 + lane) * 16);
            v16h bf = *(const v16h*)(&Bs[ks * 512 + lane * 16]);
            acc = WMMA_F16(af, bf, acc);
        }
        int ncol = lane & 15;
        #pragma unroll
        for (int r = 0; r < 8; ++r) {
            int m   = mt * 16 + r + ((lane >> 4) << 3);   // conv1 out channel
            float v = acc[r] + bias1[m];
            v = v > 0.f ? v : 0.f;
            int kk    = m * 6 + j;                        // K index for GEMM2
            int ks2   = kk >> 5;
            int kl2   = kk & 31;
            int lane2 = ((kl2 >> 4) << 4) + ncol;
            bpk[((ntile * K2S + ks2) * 32 + lane2) * 16 + (kl2 & 15)] = (_Float16)v;
        }
    }
}

// conv2 as WMMA GEMM: M=256, K=1536. Register-blocked 2x2: each workgroup
// covers two adjacent 16-wide N tiles; each wave computes 2 m-tiles x 2 n-tiles
// -> per k-step: 4 fragment loads feed 4 WMMAs (was 3 loads per WMMA).
__global__ void __launch_bounds__(256) gemm2_wmma(
        const _Float16* __restrict__ bpk, const _Float16* __restrict__ a2,
        const float* __restrict__ bias2, float* __restrict__ out) {
    int tid  = threadIdx.x;
    int ntp  = blockIdx.x;            // 512 pairs of N tiles
    int wave = tid >> 5, lane = tid & 31;
    int mt0  = wave * 2, mt1 = mt0 + 1;

    const _Float16* Bp0 = bpk + (size_t)(ntp * 2) * K2S * 512 + lane * 16;
    const _Float16* Bp1 = Bp0 + (size_t)K2S * 512;
    const _Float16* Ap0 = a2 + (size_t)mt0 * K2S * 512 + lane * 16;
    const _Float16* Ap1 = a2 + (size_t)mt1 * K2S * 512 + lane * 16;

    v8f acc00 = {}, acc01 = {}, acc10 = {}, acc11 = {};
    #pragma unroll 4
    for (int ks = 0; ks < K2S; ++ks) {
        v16h bf0 = *(const v16h*)(Bp0 + ks * 512);
        v16h bf1 = *(const v16h*)(Bp1 + ks * 512);
        v16h af0 = *(const v16h*)(Ap0 + ks * 512);
        v16h af1 = *(const v16h*)(Ap1 + ks * 512);
        acc00 = WMMA_F16(af0, bf0, acc00);
        acc01 = WMMA_F16(af0, bf1, acc01);
        acc10 = WMMA_F16(af1, bf0, acc10);
        acc11 = WMMA_F16(af1, bf1, acc11);
    }

    int ncol = lane & 15;
    int n0 = ntp * 32 + ncol;
    int n1 = n0 + 16;
    int b0 = n0 >> 9, t0 = n0 & 511;
    int b1 = n1 >> 9, t1 = n1 & 511;
    #pragma unroll
    for (int r = 0; r < 8; ++r) {
        int m0 = mt0 * 16 + r + ((lane >> 4) << 3);
        int m1 = mt1 * 16 + r + ((lane >> 4) << 3);
        float c0 = bias2[m0], c1 = bias2[m1];
        out[(size_t)(b0 * 256 + m0) * 512 + t0] = acc00[r] + c0;
        out[(size_t)(b1 * 256 + m0) * 512 + t1] = acc01[r] + c0;
        out[(size_t)(b0 * 256 + m1) * 512 + t0] = acc10[r] + c1;
        out[(size_t)(b1 * 256 + m1) * 512 + t1] = acc11[r] + c1;
    }
}

extern "C" void kernel_launch(void* const* d_in, const int* in_sizes, int n_in,
                              void* d_out, int out_size, void* d_ws, size_t ws_size,
                              hipStream_t stream) {
    (void)in_sizes; (void)n_in; (void)out_size; (void)ws_size;
    const float* x  = (const float*)d_in[0];
    const float* w1 = (const float*)d_in[1];
    const float* b1 = (const float*)d_in[2];
    const float* w2 = (const float*)d_in[3];
    const float* b2 = (const float*)d_in[4];
    float* out = (float*)d_out;

    _Float16* a1  = (_Float16*)d_ws;                           // 49,152 B
    _Float16* a2  = (_Float16*)((char*)d_ws + 65536);          // 786,432 B
    _Float16* bpk = (_Float16*)((char*)d_ws + (1u << 20));     // 50,331,648 B

    pack_w1<<<96, 256, 0, stream>>>(w1, a1);
    pack_w2<<<1536, 256, 0, stream>>>(w2, a2);
    conv1_wmma<<<6144, 256, 0, stream>>>(x, a1, b1, bpk);
    gemm2_wmma<<<512, 256, 0, stream>>>(bpk, a2, b2, out);
}
